// Attn_83777632076045
// MI455X (gfx1250) — matrix operand order repaired and checked by reading, not compiled, not run on hardware
//
#include <hip/hip_runtime.h>
#include <hip/hip_bf16.h>
#include <math.h>

#define NPTS   65536
#define DMODEL 64
#define NHEADS 8
#define HDIM   64
#define INNER  512
#define BLK    64
#define NBLK   (NPTS / BLK)

typedef __attribute__((ext_vector_type(16))) _Float16 v16h;
typedef __attribute__((ext_vector_type(8)))  _Float16 v8h;
typedef __attribute__((ext_vector_type(2)))  _Float16 v2h;
typedef __attribute__((ext_vector_type(8)))  float    v8f;

union V16U { v16h v; v8h h[2]; };

// CDNA5 WMMA: D(16x16 f32) = A(16x32 f16) x B(32x16 f16) + C   (wave32)
// A: lane L: M=L&15, half=L>>4; e -> K=16*half+e (k order chosen by this code, not the ISA table order)
//    -> one contiguous 16-half run at [kk+16*half]
// B: lane L: N=L&15, half=L>>4; e -> K=16*half+e (this code's order, same as A)  -> one contiguous 16-half run
// C: lane L: N=L&15, half=L>>4; e -> M=8*half+e
__device__ __forceinline__ v8f wmma16(v16h a, v16h b, v8f c) {
  return __builtin_amdgcn_wmma_f32_16x16x32_f16(false, a, false, b, (short)0, c,
                                                false, false);
}
__device__ __forceinline__ v16h ldA(const _Float16* row, int half, int kk) {
  V16U u;
  u.h[0] = *(const v8h*)(row + kk + 16 * half);
  u.h[1] = *(const v8h*)(row + kk + 16 * half + 8);
  return u.v;
}
// p must already include the +16*half lane offset
__device__ __forceinline__ v16h ldB(const _Float16* p) {
  V16U u;
  u.h[0] = *(const v8h*)(p);
  u.h[1] = *(const v8h*)(p + 8);
  return u.v;
}

// ---------------- weight conversion (f32 -> f16, transposed) ----------------
__global__ void cvt_f16_t_kernel(const float* __restrict__ src,
                                 _Float16* __restrict__ dst, int rows, int cols) {
  int i = blockIdx.x * blockDim.x + threadIdx.x;
  if (i < rows * cols) {
    int r = i / cols, c = i % cols;
    dst[(size_t)c * rows + r] = (_Float16)src[i];  // dst: cols x rows
  }
}

__global__ void rpe_prep_kernel(const float* __restrict__ w_rpe_w,
                                const float* __restrict__ w_rpe_b,
                                float* __restrict__ weff, float* __restrict__ bh) {
  int t = threadIdx.x;
  if (t < 16) {
    int c = t >> 3, h = t & 7;
    float s = 0.f;
    for (int w = 0; w < 8; ++w)
      for (int dd = 0; dd < 64; ++dd)
        s += fmaxf(w_rpe_w[(w * 2 + c) * INNER + h * 64 + dd], 0.f);
    weff[c * 8 + h] = s;
  } else if (t < 24) {
    int h = t - 16;
    float s = 0.f;
    for (int dd = 0; dd < 64; ++dd) s += w_rpe_b[h * 64 + dd];
    bh[h] = s;
  }
}

// ---------------- LayerNorm: one wave per 64-wide row ----------------
__global__ __launch_bounds__(256) void layernorm_f16_kernel(
    const float* __restrict__ x, const float* __restrict__ g,
    const float* __restrict__ b, _Float16* __restrict__ y) {
  int n = blockIdx.x * 8 + (threadIdx.x >> 5);
  int lane = threadIdx.x & 31;
  const float* xr = x + (size_t)n * DMODEL;
  float v0 = xr[lane], v1 = xr[lane + 32];
  float s = v0 + v1;
  #pragma unroll
  for (int off = 16; off; off >>= 1) s += __shfl_xor(s, off, 32);
  float mu = s * (1.f / 64.f);
  float d0 = v0 - mu, d1 = v1 - mu;
  float var = d0 * d0 + d1 * d1;
  #pragma unroll
  for (int off = 16; off; off >>= 1) var += __shfl_xor(var, off, 32);
  float inv = rsqrtf(var * (1.f / 64.f) + 1e-5f);
  _Float16* yr = y + (size_t)n * DMODEL;
  yr[lane]      = (_Float16)(d0 * inv * g[lane] + b[lane]);
  yr[lane + 32] = (_Float16)(d1 * inv * g[lane + 32] + b[lane + 32]);
}

// ---------------- q/k/v = xn @ W, Wt pre-transposed (512 x 64) ----------------
__global__ __launch_bounds__(256) void qkv_gemm_kernel(
    const _Float16* __restrict__ xn, const _Float16* __restrict__ wqt,
    const _Float16* __restrict__ wkt, const _Float16* __restrict__ wvt,
    _Float16* __restrict__ q, _Float16* __restrict__ k, _Float16* __restrict__ v) {
  const int wave = threadIdx.x >> 5;
  const int lane = threadIdx.x & 31;
  const int half = lane >> 4, lcol = lane & 15;
  const int mrow = blockIdx.x * 16;
  const _Float16* Wt = (blockIdx.z == 0) ? wqt : (blockIdx.z == 1) ? wkt : wvt;
  _Float16* O        = (blockIdx.z == 0) ? q   : (blockIdx.z == 1) ? k   : v;

  const _Float16* arow = xn + (size_t)(mrow + lcol) * DMODEL;
  v16h a0 = ldA(arow, half, 0);
  v16h a1 = ldA(arow, half, 32);
  const int nbase = wave * 64;
  for (int nt = 0; nt < 4; ++nt) {
    const int ncol = nbase + nt * 16 + lcol;
    const _Float16* wrow = Wt + (size_t)ncol * DMODEL + 16 * half;
    v16h b0 = ldB(wrow);
    v16h b1 = ldB(wrow + 32);
    v8f c = {};
    c = wmma16(a0, b0, c);
    c = wmma16(a1, b1, c);
    #pragma unroll
    for (int e = 0; e < 8; ++e)
      O[(size_t)(mrow + half * 8 + e) * INNER + ncol] = (_Float16)c[e];
  }
}

// ---------------- LSH codes, one wave per row (v8h loads) ----------------
__global__ __launch_bounds__(256) void lsh_codes_kernel(
    const _Float16* __restrict__ kbuf, const float* __restrict__ coords,
    const float* __restrict__ lsh_a, float* __restrict__ keys0,
    int* __restrict__ idx0, float* __restrict__ keys1, int* __restrict__ idx1) {
  int n = blockIdx.x * 8 + (threadIdx.x >> 5);
  int lane = threadIdx.x & 31;
  const _Float16* kr = kbuf + (size_t)n * INNER;
  float s0 = 0.f, s1 = 0.f, nrm = 0.f;
  #pragma unroll
  for (int cch = 0; cch < 2; ++cch) {
    int j0 = lane * 8 + cch * 256;
    v8h kv8 = *(const v8h*)(kr + j0);
    #pragma unroll
    for (int t = 0; t < 8; ++t) {
      float kv = (float)kv8[t];
      nrm += kv * kv;
      s0 += kv * lsh_a[(j0 + t) * 2 + 0];
      s1 += kv * lsh_a[(j0 + t) * 2 + 1];
    }
  }
  #pragma unroll
  for (int off = 16; off; off >>= 1) {
    nrm += __shfl_xor(nrm, off, 32);
    s0  += __shfl_xor(s0, off, 32);
    s1  += __shfl_xor(s1, off, 32);
  }
  if (lane == 0) {
    float inv = 1.f / (sqrtf(nrm) + 1e-6f);
    float c0 = s0 * inv, c1 = s1 * inv;
    for (int c = 0; c < 3; ++c) {
      float cv = coords[(size_t)n * 3 + c];
      c0 += cv * lsh_a[(INNER + c) * 2 + 0];
      c1 += cv * lsh_a[(INNER + c) * 2 + 1];
    }
    keys0[n] = c0; idx0[n] = n;
    keys1[n] = c1; idx1[n] = n;
  }
}

__global__ void bitonic_step_kernel(float* __restrict__ keys,
                                    int* __restrict__ idx, unsigned j, unsigned k) {
  unsigned i = blockIdx.x * blockDim.x + threadIdx.x;
  unsigned p = i ^ j;
  if (p > i) {
    float ki = keys[i], kp = keys[p];
    bool asc = ((i & k) == 0);
    if ((ki > kp) == asc) {
      keys[i] = kp; keys[p] = ki;
      int t = idx[i]; idx[i] = idx[p]; idx[p] = t;
    }
  }
}

// ---------------- blocked attention: one wave per (sorted block, head) -------
// LDS: S 64x66 f32 | P16 64x72 f16 | Vt 64x72 f16 | ps 128 f32 | perm 64 i32
#define S_STRIDE  66
#define H_STRIDE  72
__global__ __launch_bounds__(32) void attn_block_kernel(
    const _Float16* __restrict__ qbuf, const _Float16* __restrict__ kbuf,
    const _Float16* __restrict__ vbuf, const int* __restrict__ perm,
    const float* __restrict__ coords, const float* __restrict__ weff,
    const float* __restrict__ bh, _Float16* __restrict__ obuf,
    float* __restrict__ lsebuf) {
  extern __shared__ char smemraw[];
  float*    S   = (float*)smemraw;                        // 64*66 f32
  _Float16* P16 = (_Float16*)(S + 64 * S_STRIDE);         // 64*72 f16
  _Float16* Vt  = P16 + 64 * H_STRIDE;                    // 64*72 f16
  float*    psS = (float*)(Vt + 64 * H_STRIDE);           // 128 f32
  int*    permS = (int*)(psS + 128);                      // 64 i32

  const int b = blockIdx.x;
  const int h = blockIdx.y;
  const int lane = threadIdx.x;
  const int half = lane >> 4, lcol = lane & 15;

  for (int i = lane; i < 64; i += 32) {
    int p = perm[b * 64 + i];
    permS[i] = p;
    psS[i * 2 + 0] = coords[(size_t)p * 3 + 1];
    psS[i * 2 + 1] = coords[(size_t)p * 3 + 2];
  }
  __syncthreads();

  // stage V tile transposed: Vt[d][j] = V[perm[j]][h*64+d]
  {
    const int dgrp = (lane & 7) * 8;
    const int joff = (lane >> 3) * 2;
    for (int j0 = 0; j0 < 64; j0 += 8) {
      int j = j0 + joff;
      v8h r0 = *(const v8h*)(vbuf + (size_t)permS[j] * INNER + h * HDIM + dgrp);
      v8h r1 = *(const v8h*)(vbuf + (size_t)permS[j + 1] * INNER + h * HDIM + dgrp);
      #pragma unroll
      for (int t = 0; t < 8; ++t) {
        v2h pr;
        pr[0] = r0[t];
        pr[1] = r1[t];
        *(v2h*)(Vt + (size_t)(dgrp + t) * H_STRIDE + j) = pr;
      }
    }
  }

  const float scale = 0.125f;
  const float we0 = weff[0 * 8 + h], we1 = weff[1 * 8 + h], bhh = bh[h];

  // S = Q K^T * scale - rpe - b_h
  for (int mt = 0; mt < 4; ++mt) {
    const _Float16* qrow =
        qbuf + (size_t)permS[mt * 16 + lcol] * INNER + h * HDIM;
    v16h a0 = ldA(qrow, half, 0);
    v16h a1 = ldA(qrow, half, 32);
    for (int nt = 0; nt < 4; ++nt) {
      const _Float16* krow =
          kbuf + (size_t)permS[nt * 16 + lcol] * INNER + h * HDIM + 16 * half;
      v16h b0 = ldB(krow);
      v16h b1 = ldB(krow + 32);
      v8f c = {};
      c = wmma16(a0, b0, c);
      c = wmma16(a1, b1, c);
      #pragma unroll
      for (int e = 0; e < 8; ++e) {
        int r = mt * 16 + half * 8 + e, cc = nt * 16 + lcol;
        float d0 = psS[r * 2 + 0] - psS[cc * 2 + 0];
        float d1 = psS[r * 2 + 1] - psS[cc * 2 + 1];
        S[r * S_STRIDE + cc] =
            c[e] * scale - (we0 * d0 * d0 + we1 * d1 * d1) - bhh;
      }
    }
  }

  // row softmax -> P16 (f16, A-fragment friendly), lse scatter
  for (int r = lane; r < 64; r += 32) {
    const float2* row = (const float2*)(S + r * S_STRIDE);
    float m = -3.4e38f;
    #pragma unroll 8
    for (int j = 0; j < 32; ++j) {
      float2 t = row[j];
      m = fmaxf(m, fmaxf(t.x, t.y));
    }
    float s = 0.f;
    #pragma unroll 8
    for (int j = 0; j < 32; ++j) {
      float2 t = row[j];
      s += __expf(t.x - m) + __expf(t.y - m);
    }
    float lse = m + __logf(s);
    for (int j0 = 0; j0 < 64; j0 += 8) {
      v8h pv;
      #pragma unroll
      for (int t = 0; t < 8; ++t)
        pv[t] = (_Float16)__expf(S[r * S_STRIDE + j0 + t] - lse);
      *(v8h*)(P16 + (size_t)r * H_STRIDE + j0) = pv;
    }
    lsebuf[(size_t)permS[r] * NHEADS + h] = lse;
  }

  // O = P V, scatter through perm
  for (int mt = 0; mt < 4; ++mt) {
    const _Float16* prow = P16 + (size_t)(mt * 16 + lcol) * H_STRIDE;
    v16h pa0 = ldA(prow, half, 0);
    v16h pa1 = ldA(prow, half, 32);
    for (int nt = 0; nt < 4; ++nt) {
      const _Float16* vrow =
          Vt + (size_t)(nt * 16 + lcol) * H_STRIDE + 16 * half;
      v16h b0 = ldB(vrow);
      v16h b1 = ldB(vrow + 32);
      v8f c = {};
      c = wmma16(pa0, b0, c);
      c = wmma16(pa1, b1, c);
      #pragma unroll
      for (int e = 0; e < 8; ++e) {
        int r = mt * 16 + half * 8 + e;
        obuf[(size_t)permS[r] * INNER + h * HDIM + nt * 16 + lcol] =
            (_Float16)c[e];
      }
    }
  }
}

// ---------------- combine two hash rounds (v8h vectorized) ----------------
__global__ void combine_kernel(const _Float16* __restrict__ o0,
                               const _Float16* __restrict__ o1,
                               const float* __restrict__ lse0,
                               const float* __restrict__ lse1,
                               _Float16* __restrict__ out) {
  size_t chunk = (size_t)blockIdx.x * blockDim.x + threadIdx.x;
  size_t base = chunk * 8;
  size_t n = base >> 9;
  int c = (int)(base & 511);
  int h = c >> 6;
  float l0 = lse0[n * NHEADS + h], l1 = lse1[n * NHEADS + h];
  float m = fmaxf(l0, l1);
  float e0 = __expf(l0 - m), e1 = __expf(l1 - m);
  float inv = 1.f / (e0 + e1);
  v8h a = *(const v8h*)(o0 + base);
  v8h bq = *(const v8h*)(o1 + base);
  v8h o;
  #pragma unroll
  for (int t = 0; t < 8; ++t)
    o[t] = (_Float16)(((float)a[t] * e0 + (float)bq[t] * e1) * inv);
  *(v8h*)(out + base) = o;
}

// ---------------- out projection + residual; owt pre-transposed (64x512) ----
__global__ __launch_bounds__(128) void out_proj_kernel(
    const _Float16* __restrict__ comb, const _Float16* __restrict__ owt,
    const float* __restrict__ outb, const float* __restrict__ x,
    float* __restrict__ xo) {
  const int wave = threadIdx.x >> 5;
  const int lane = threadIdx.x & 31;
  const int half = lane >> 4, lcol = lane & 15;
  const int mrow = blockIdx.x * 16;
  const int ncol = wave * 16 + lcol;
  const _Float16* arow = comb + (size_t)(mrow + lcol) * INNER;
  const _Float16* wrow = owt + (size_t)ncol * INNER + 16 * half;
  v8f c = {};
  for (int kk = 0; kk < INNER; kk += 32) {
    v16h a = ldA(arow, half, kk);
    v16h bv = ldB(wrow + kk);
    c = wmma16(a, bv, c);
  }
  #pragma unroll
  for (int e = 0; e < 8; ++e) {
    size_t o = (size_t)(mrow + half * 8 + e) * DMODEL + ncol;
    xo[o] = x[o] + c[e] + outb[ncol];
  }
}

// ---------------- FFN; w1t/w2t pre-transposed (64x64) ----------------
__global__ __launch_bounds__(256) void ffn_kernel(
    const _Float16* __restrict__ h16, const _Float16* __restrict__ w1t,
    const float* __restrict__ b1, const _Float16* __restrict__ w2t,
    const float* __restrict__ b2, float* __restrict__ xo) {
  __shared__ _Float16 tbuf[8][16 * H_STRIDE];
  const int wave = threadIdx.x >> 5;
  const int lane = threadIdx.x & 31;
  const int half = lane >> 4, lcol = lane & 15;
  const int mrow = blockIdx.x * 128 + wave * 16;
  _Float16* t16 = tbuf[wave];

  const _Float16* arow = h16 + (size_t)(mrow + lcol) * DMODEL;
  v16h a0 = ldA(arow, half, 0);
  v16h a1 = ldA(arow, half, 32);
  for (int nt = 0; nt < 4; ++nt) {
    const int ncol = nt * 16 + lcol;
    const _Float16* wrow = w1t + (size_t)ncol * DMODEL + 16 * half;
    v16h b0 = ldB(wrow);
    v16h b1 = ldB(wrow + 32);
    v8f c = {};
    c = wmma16(a0, b0, c);
    c = wmma16(a1, b1, c);
    #pragma unroll
    for (int e = 0; e < 8; ++e)
      t16[(half * 8 + e) * H_STRIDE + ncol] =
          (_Float16)fmaxf(c[e] + b1[ncol], 0.f);
  }

  const _Float16* prow = t16 + (size_t)lcol * H_STRIDE;
  v16h pa0 = ldA(prow, half, 0);
  v16h pa1 = ldA(prow, half, 32);
  for (int nt = 0; nt < 4; ++nt) {
    const int ncol = nt * 16 + lcol;
    const _Float16* wrow = w2t + (size_t)ncol * DMODEL + 16 * half;
    v16h b0 = ldB(wrow);
    v16h b1 = ldB(wrow + 32);
    v8f c = {};
    c = wmma16(pa0, b0, c);
    c = wmma16(pa1, b1, c);
    #pragma unroll
    for (int e = 0; e < 8; ++e) {
      size_t o = (size_t)(mrow + half * 8 + e) * DMODEL + ncol;
      xo[o] = xo[o] + c[e] + b2[ncol];
    }
  }
}

extern "C" void kernel_launch(void* const* d_in, const int* in_sizes, int n_in,
                              void* d_out, int out_size, void* d_ws,
                              size_t ws_size, hipStream_t stream) {
  (void)in_sizes; (void)n_in; (void)out_size; (void)ws_size;
  const float* x       = (const float*)d_in[0];
  const float* coords  = (const float*)d_in[1];
  const float* w_q     = (const float*)d_in[2];
  const float* w_k     = (const float*)d_in[3];
  const float* w_v     = (const float*)d_in[4];
  const float* ln1_g   = (const float*)d_in[5];
  const float* ln1_b   = (const float*)d_in[6];
  const float* ln2_g   = (const float*)d_in[7];
  const float* ln2_b   = (const float*)d_in[8];
  const float* ff_w1   = (const float*)d_in[9];
  const float* ff_b1   = (const float*)d_in[10];
  const float* ff_w2   = (const float*)d_in[11];
  const float* ff_b2   = (const float*)d_in[12];
  const float* w_rpe_w = (const float*)d_in[13];
  const float* w_rpe_b = (const float*)d_in[14];
  const float* out_w   = (const float*)d_in[15];
  const float* out_b   = (const float*)d_in[16];
  const float* lsh_a   = (const float*)d_in[17];
  float* xo = (float*)d_out;

  char* ws = (char*)d_ws;
  size_t off = 0;
  auto alloc = [&](size_t bytes) -> void* {
    void* p = ws + off;
    off += (bytes + 255) & ~(size_t)255;
    return p;
  };
  _Float16* xn16 = (_Float16*)alloc((size_t)NPTS * DMODEL * 2);
  _Float16* q16  = (_Float16*)alloc((size_t)NPTS * INNER * 2);
  _Float16* k16  = (_Float16*)alloc((size_t)NPTS * INNER * 2);
  _Float16* v16  = (_Float16*)alloc((size_t)NPTS * INNER * 2);
  _Float16* o0   = (_Float16*)alloc((size_t)NPTS * INNER * 2);
  _Float16* o1   = (_Float16*)alloc((size_t)NPTS * INNER * 2);
  _Float16* wqt  = (_Float16*)alloc((size_t)DMODEL * INNER * 2);  // 512x64
  _Float16* wkt  = (_Float16*)alloc((size_t)DMODEL * INNER * 2);
  _Float16* wvt  = (_Float16*)alloc((size_t)DMODEL * INNER * 2);
  _Float16* owt  = (_Float16*)alloc((size_t)INNER * DMODEL * 2);  // 64x512
  _Float16* fw1t = (_Float16*)alloc((size_t)DMODEL * DMODEL * 2);
  _Float16* fw2t = (_Float16*)alloc((size_t)DMODEL * DMODEL * 2);
  float* lse0  = (float*)alloc((size_t)NPTS * NHEADS * 4);
  float* lse1  = (float*)alloc((size_t)NPTS * NHEADS * 4);
  float* keys0 = (float*)alloc((size_t)NPTS * 4);
  float* keys1 = (float*)alloc((size_t)NPTS * 4);
  int*   idx0  = (int*)alloc((size_t)NPTS * 4);
  int*   idx1  = (int*)alloc((size_t)NPTS * 4);
  float* weff  = (float*)alloc(16 * 4);
  float* bh    = (float*)alloc(8 * 4);

  const int nw = DMODEL * INNER;
  cvt_f16_t_kernel<<<(nw + 255) / 256, 256, 0, stream>>>(w_q, wqt, DMODEL, INNER);
  cvt_f16_t_kernel<<<(nw + 255) / 256, 256, 0, stream>>>(w_k, wkt, DMODEL, INNER);
  cvt_f16_t_kernel<<<(nw + 255) / 256, 256, 0, stream>>>(w_v, wvt, DMODEL, INNER);
  cvt_f16_t_kernel<<<(nw + 255) / 256, 256, 0, stream>>>(out_w, owt, INNER, DMODEL);
  cvt_f16_t_kernel<<<(DMODEL * DMODEL + 255) / 256, 256, 0, stream>>>(ff_w1, fw1t, DMODEL, DMODEL);
  cvt_f16_t_kernel<<<(DMODEL * DMODEL + 255) / 256, 256, 0, stream>>>(ff_w2, fw2t, DMODEL, DMODEL);
  rpe_prep_kernel<<<1, 64, 0, stream>>>(w_rpe_w, w_rpe_b, weff, bh);

  layernorm_f16_kernel<<<NPTS / 8, 256, 0, stream>>>(x, ln1_g, ln1_b, xn16);

  qkv_gemm_kernel<<<dim3(NPTS / 16, 1, 3), 256, 0, stream>>>(xn16, wqt, wkt,
                                                             wvt, q16, k16, v16);

  lsh_codes_kernel<<<NPTS / 8, 256, 0, stream>>>(k16, coords, lsh_a, keys0,
                                                 idx0, keys1, idx1);

  for (int r = 0; r < 2; ++r) {
    float* keys = r ? keys1 : keys0;
    int* idx = r ? idx1 : idx0;
    for (unsigned k = 2; k <= NPTS; k <<= 1)
      for (unsigned j = k >> 1; j >= 1; j >>= 1)
        bitonic_step_kernel<<<NPTS / 256, 256, 0, stream>>>(keys, idx, j, k);
  }

  const size_t attn_smem = (size_t)64 * S_STRIDE * 4 +
                           (size_t)64 * H_STRIDE * 2 * 2 + 128 * 4 + 64 * 4;
  attn_block_kernel<<<dim3(NBLK, NHEADS), 32, attn_smem, stream>>>(
      q16, k16, v16, idx0, coords, weff, bh, o0, lse0);
  attn_block_kernel<<<dim3(NBLK, NHEADS), 32, attn_smem, stream>>>(
      q16, k16, v16, idx1, coords, weff, bh, o1, lse1);

  _Float16* comb = q16;
  combine_kernel<<<(unsigned)((size_t)NPTS * INNER / 8 / 256), 256, 0, stream>>>(
      o0, o1, lse0, lse1, comb);

  out_proj_kernel<<<NPTS / 16, 128, 0, stream>>>(comb, owt, out_b, x, xo);

  layernorm_f16_kernel<<<NPTS / 8, 256, 0, stream>>>(xo, ln2_g, ln2_b, xn16);
  ffn_kernel<<<NPTS / 128, 256, 0, stream>>>(xn16, fw1t, ff_b1, fw2t, ff_b2, xo);
}
